// TensorConvolutionTrainLayer_23072564314661
// MI455X (gfx1250) — compile-verified, matrix operand order checked by reading, not run on hardware
//
#include <hip/hip_runtime.h>

typedef __attribute__((ext_vector_type(16))) _Float16 v16h;
typedef __attribute__((ext_vector_type(8)))  _Float16 v8h;
typedef __attribute__((ext_vector_type(8)))  float    v8f;

#define BATCH   2048
#define PATCHES 196
#define PIXELS  256
#define NCAR    8
#define BOND    16
#define OUTC    10
#define NROWS   (BATCH * PATCHES)   /* 401408 */
#define NTILES  (NROWS / 16)        /* 25088 exact */
#define NKC     7                   /* K chunks of 32 over padded 224 patches */
#define NSTILE  (BATCH / 16)        /* 128 batch tiles */
#define FRAG    512                 /* elems per 16x32 fragment (32 lanes x 16 halves) */

// Fragment addressing (CDNA5 ISA 7.12.2, 16-bit A/B layouts):
// lane<16  -> row/col = lane,    K in {0..7, 16..23}
// lane>=16 -> row/col = lane-16, K in {8..15, 24..31}
__device__ __host__ inline int frag_slot(int kr /*0..31*/, int m /*0..15*/, int* lane_out)
{
    int base    = (kr & 8) ? 8 : 0;
    int laneoff = (kr & 8) ? 16 : 0;
    int j = (kr < 16) ? (kr - base) : (kr - base - 8);
    *lane_out = m + laneoff;
    return j;
}

__device__ inline v16h load_frag(const _Float16* p, int lane)
{
    return *(const v16h*)(p + lane * 16);           // 32B/lane, coalesced b128 x2
}

__device__ inline v8f wmma_f16(v16h a, v16h b, v8f c)
{
    return __builtin_amdgcn_wmma_f32_16x16x32_f16(
        false, a, false, b, (short)0, c, false, false);
}

// ---------------------------------------------------------------------------
// Pack conv_w into 8 B-fragments [kc 8][512]: B[k][n] = conv_w[n][k], n<8 else 0
// ---------------------------------------------------------------------------
__global__ void pack_cw_kernel(const float* __restrict__ cw, _Float16* __restrict__ B)
{
    int idx = blockIdx.x * 256 + threadIdx.x;       // 8 * 512
    if (idx >= 8 * FRAG) return;
    int j = idx & 15, lane = (idx >> 4) & 31, kc = idx >> 9;
    int base = (lane < 16) ? 0 : 8;
    int k = kc * 32 + base + ((j < 8) ? j : (j + 8));
    int n = lane & 15;
    float v = (n < NCAR) ? cw[n * PIXELS + k] : 0.f;
    B[idx] = (_Float16)v;
}

// ---------------------------------------------------------------------------
// Stage 1 (WMMA): y[n,(s,q)] = sum_p x[(s,q),p] * conv_w[n,p].
// One 16-row x 256-pixel tile per wave iteration; f32->f16 convert lands
// directly in A-fragment layout in a private 4KB LDS slab; 8 WMMAs per tile.
// D written (f32->f16) into yA fragment layout consumed by the chain kernel.
// ---------------------------------------------------------------------------
__global__ __launch_bounds__(256) void conv_y_kernel(
    const float* __restrict__ x, const _Float16* __restrict__ Bcw,
    _Float16* __restrict__ yA)
{
    __shared__ _Float16 slab[8][4 * FRAG];          // 4KB per wave, 32KB total

    const int lane = threadIdx.x & 31;
    const int wave = threadIdx.x >> 5;
    const int gw = blockIdx.x * 8 + wave;
    const int nwaves = gridDim.x * 8;
    const int mrow = (lane < 16) ? 0 : 8;
    const int ncol = lane & 15;

    // preload the 8 conv_w B-fragments into registers (64 VGPRs)
    v16h bw[8];
#pragma unroll
    for (int kc = 0; kc < 8; ++kc)
        bw[kc] = load_frag(Bcw + kc * FRAG, lane);

    for (int tile = gw; tile < NTILES; tile += nwaves) {
        const int row0 = tile * 16;
        v8f acc = {};
#pragma unroll
        for (int h = 0; h < 2; ++h) {
            // ---- phase A: load+convert 16 rows x 128 px into fragment slabs
#pragma unroll
            for (int t = 0; t < 8; ++t) {
                int gidx = t * 32 + lane;           // 0..255
                int row  = gidx >> 4;               // 0..15
                int pcl  = gidx & 15;               // 8-float chunk within half
                int pc   = h * 16 + pcl;            // global 8-float chunk
                const float4* g = (const float4*)(x + (size_t)(row0 + row) * PIXELS + pc * 8);
                float4 u0 = g[0];
                float4 u1 = g[1];
                v8h hv;
                hv[0] = (_Float16)u0.x; hv[1] = (_Float16)u0.y;
                hv[2] = (_Float16)u0.z; hv[3] = (_Float16)u0.w;
                hv[4] = (_Float16)u1.x; hv[5] = (_Float16)u1.y;
                hv[6] = (_Float16)u1.z; hv[7] = (_Float16)u1.w;
                int kcl = pc >> 2;                  // frag within slab would be (pc>>2)
                kcl &= 3;                           // 4 frags per half
                int sub = pc & 3;                   // 8-k subchunk within frag
                int slot = (row + ((sub & 1) ? 16 : 0)) * 16 + ((sub & 2) ? 8 : 0);
                *(v8h*)&slab[wave][kcl * FRAG + slot] = hv;
            }
            // ---- phase B: 4 WMMAs on the staged fragments
#pragma unroll
            for (int kcl = 0; kcl < 4; ++kcl) {
                v16h a = *(const v16h*)&slab[wave][kcl * FRAG + lane * 16];
                acc = wmma_f16(a, bw[h * 4 + kcl], acc);
            }
        }
        // ---- epilogue: scatter D (carriage = ncol) into yA fragment layout
        if (ncol < NCAR) {
#pragma unroll
            for (int i = 0; i < 8; ++i) {
                int row = row0 + i + mrow;
                int s = row / PATCHES;
                int q = row - s * PATCHES;
                int stile = s >> 4, m = s & 15;
                int kc = q >> 5, kr = q & 31;
                int fl;
                int j = frag_slot(kr, m, &fl);
                size_t addr = ((size_t)((ncol * NSTILE + stile) * NKC + kc)) * FRAG + fl * 16 + j;
                yA[addr] = (_Float16)acc[i];
            }
        }
    }
}

// Zero-fill the K padding slots (q = 196..223, kc = 6)
__global__ void pad_y_kernel(_Float16* __restrict__ yA)
{
    int idx = blockIdx.x * 256 + threadIdx.x;       // 8 * 128 * 16 * 28
    if (idx >= NCAR * NSTILE * 16 * 28) return;
    int kk = idx % 28; int t = idx / 28;
    int m = t & 15; t >>= 4;
    int stile = t & (NSTILE - 1); int n = t / NSTILE;
    int kr = 4 + kk;                                // 4..31 within kc=6
    int fl;
    int j = frag_slot(kr, m, &fl);
    size_t addr = ((size_t)((n * NSTILE + stile) * NKC + 6)) * FRAG + fl * 16 + j;
    yA[addr] = (_Float16)0.f;
}

// ---------------------------------------------------------------------------
// Pack kernels: build B factors directly in fragment layout (zero K-padded)
// ---------------------------------------------------------------------------
__global__ void pack_first_kernel(const float* __restrict__ A, _Float16* __restrict__ B)
{
    int idx = blockIdx.x * 256 + threadIdx.x;       // [7][10] frags
    if (idx >= NKC * OUTC * FRAG) return;
    int j = idx & 15, lane = (idx >> 4) & 31, frag = idx >> 9;
    int kc = frag / OUTC, t = frag - kc * OUTC;     // t = c
    int base = (lane < 16) ? 0 : 8;
    int k = kc * 32 + base + ((j < 8) ? j : (j + 8));
    int r = lane & 15;
    float v = (k < PATCHES) ? A[((size_t)t * PATCHES + k) * BOND + r] : 0.f;
    B[idx] = (_Float16)v;
}

__global__ void pack_mid_kernel(const float* __restrict__ A, _Float16* __restrict__ B)
{
    int idx = blockIdx.x * 256 + threadIdx.x;       // [6][7][16] frags
    if (idx >= 6 * NKC * 16 * FRAG) return;
    int j = idx & 15, lane = (idx >> 4) & 31, frag = idx >> 9;
    int i = frag / (NKC * 16);
    int rem = frag - i * (NKC * 16);
    int kc = rem >> 4, t = rem & 15;                // t = r
    int base = (lane < 16) ? 0 : 8;
    int k = kc * 32 + base + ((j < 8) ? j : (j + 8));
    int w = lane & 15;
    float v = (k < PATCHES) ? A[(((size_t)(i * BOND + t) * PATCHES) + k) * BOND + w] : 0.f;
    B[idx] = (_Float16)v;
}

__global__ void pack_last_kernel(const float* __restrict__ A, _Float16* __restrict__ B)
{
    int idx = blockIdx.x * 256 + threadIdx.x;       // [7] frags
    if (idx >= NKC * FRAG) return;
    int j = idx & 15, lane = (idx >> 4) & 31, kc = idx >> 9;
    int base = (lane < 16) ? 0 : 8;
    int k = kc * 32 + base + ((j < 8) ? j : (j + 8));
    int r = lane & 15;
    float v = (k < PATCHES) ? A[(size_t)r * PATCHES + k] : 0.f;
    B[idx] = (_Float16)v;
}

// ---------------------------------------------------------------------------
// Stage 2: per-16-batch-row tile, MPS chain; A/B fragments straight from
// global (L2-resident), LDS only for the per-batch chain updates.
// ---------------------------------------------------------------------------
__global__ __launch_bounds__(256) void chain_kernel(
    const _Float16* __restrict__ yA,   // [8][128][7][512]
    const _Float16* __restrict__ Bf,   // [7][10][512]
    const _Float16* __restrict__ Bm,   // [6][7][16][512]
    const _Float16* __restrict__ Bl,   // [7][512]
    float* __restrict__ out)
{
    __shared__ float Gs[16 * 256];
    __shared__ float Ms[2][16 * 160];
    __shared__ float Vs[16 * 16];

    const int tid  = threadIdx.x;
    const int lane = tid & 31;
    const int wave = tid >> 5;
    const int stile = blockIdx.x;
    const int mrow = (lane < 16) ? 0 : 8;
    const int col  = lane & 15;

    // ---- first carriage: M0[s, c*16+r]
    v8f aF0 = {}; v8f aF1 = {};
#pragma unroll
    for (int kc = 0; kc < NKC; ++kc) {
        v16h a = load_frag(yA + ((size_t)((0 * NSTILE + stile) * NKC + kc) << 9), lane);
        const _Float16* bb = Bf + ((size_t)(kc * OUTC) << 9);
        aF0 = wmma_f16(a, load_frag(bb + ((size_t)wave << 9), lane), aF0);
        if (wave < 2)
            aF1 = wmma_f16(a, load_frag(bb + ((size_t)(wave + 8) << 9), lane), aF1);
    }
#pragma unroll
    for (int i = 0; i < 8; ++i)
        Ms[0][(i + mrow) * 160 + wave * 16 + col] = aF0[i];
    if (wave < 2) {
#pragma unroll
        for (int i = 0; i < 8; ++i)
            Ms[0][(i + mrow) * 160 + (wave + 8) * 16 + col] = aF1[i];
    }
    __syncthreads();

    // ---- middle carriages
    int cur = 0;
    for (int ic = 0; ic < 6; ++ic) {
        v8f c0 = {}; v8f c1 = {};
#pragma unroll
        for (int kc = 0; kc < NKC; ++kc) {
            v16h a = load_frag(yA + ((size_t)(((ic + 1) * NSTILE + stile) * NKC + kc) << 9), lane);
            const _Float16* bb = Bm + ((size_t)((ic * NKC + kc) * 16) << 9);
            c0 = wmma_f16(a, load_frag(bb + ((size_t)wave << 9), lane), c0);
            c1 = wmma_f16(a, load_frag(bb + ((size_t)(wave + 8) << 9), lane), c1);
        }
#pragma unroll
        for (int i = 0; i < 8; ++i) {
            int m = i + mrow;
            Gs[m * 256 + wave * 16 + col]       = c0[i];
            Gs[m * 256 + (wave + 8) * 16 + col] = c1[i];
        }
        __syncthreads();
        for (int o = tid; o < 16 * 160; o += 256) {
            int s = o / 160, rem = o - s * 160;
            int c = rem >> 4, w = rem & 15;
            float sum = 0.f;
#pragma unroll
            for (int r = 0; r < 16; ++r)
                sum += Ms[cur][s * 160 + c * 16 + r] * Gs[s * 256 + r * 16 + w];
            Ms[cur ^ 1][s * 160 + c * 16 + w] = sum;
        }
        cur ^= 1;
        __syncthreads();
    }

    // ---- last carriage: v[s,r] = sum_q Y7[s,q] * Blast[q,r]
    if (wave == 0) {
        v8f av = {};
#pragma unroll
        for (int kc = 0; kc < NKC; ++kc) {
            v16h a = load_frag(yA + ((size_t)((7 * NSTILE + stile) * NKC + kc) << 9), lane);
            av = wmma_f16(a, load_frag(Bl + ((size_t)kc << 9), lane), av);
        }
#pragma unroll
        for (int i = 0; i < 8; ++i)
            Vs[(i + mrow) * 16 + col] = av[i];
    }
    __syncthreads();

    // ---- out[s,c] = sum_r M[s,c,r] * v[s,r]
    for (int o = tid; o < 16 * OUTC; o += 256) {
        int s = o / OUTC, c = o - s * OUTC;
        float sum = 0.f;
#pragma unroll
        for (int r = 0; r < 16; ++r)
            sum += Ms[cur][s * 160 + c * 16 + r] * Vs[s * 16 + r];
        out[(size_t)(stile * 16 + s) * OUTC + c] = sum;
    }
}

// ---------------------------------------------------------------------------
extern "C" void kernel_launch(void* const* d_in, const int* in_sizes, int n_in,
                              void* d_out, int out_size, void* d_ws, size_t ws_size,
                              hipStream_t stream)
{
    const float* x       = (const float*)d_in[0];
    const float* conv_w  = (const float*)d_in[1];
    const float* A_first = (const float*)d_in[2];
    const float* A_mid   = (const float*)d_in[3];
    const float* A_last  = (const float*)d_in[4];
    float* out = (float*)d_out;

    char* ws = (char*)d_ws;
    size_t off = 0;
    _Float16* yA = (_Float16*)(ws + off);
    off += (size_t)NCAR * NSTILE * NKC * FRAG * 2;          // 7.34 MB
    off = (off + 255) & ~(size_t)255;
    _Float16* Bf = (_Float16*)(ws + off);
    off += (size_t)NKC * OUTC * FRAG * 2;
    off = (off + 255) & ~(size_t)255;
    _Float16* Bm = (_Float16*)(ws + off);
    off += (size_t)6 * NKC * 16 * FRAG * 2;
    off = (off + 255) & ~(size_t)255;
    _Float16* Bl = (_Float16*)(ws + off);
    off += (size_t)NKC * FRAG * 2;
    off = (off + 255) & ~(size_t)255;
    _Float16* Bcw = (_Float16*)(ws + off);
    off += (size_t)8 * FRAG * 2;
    (void)ws_size; (void)in_sizes; (void)n_in; (void)out_size;

    pack_cw_kernel<<<(8 * FRAG + 255) / 256, 256, 0, stream>>>(conv_w, Bcw);
    conv_y_kernel<<<1024, 256, 0, stream>>>(x, Bcw, yA);
    pad_y_kernel<<<(NCAR * NSTILE * 16 * 28 + 255) / 256, 256, 0, stream>>>(yA);
    pack_first_kernel<<<(NKC * OUTC * FRAG + 255) / 256, 256, 0, stream>>>(A_first, Bf);
    pack_mid_kernel<<<(6 * NKC * 16 * FRAG + 255) / 256, 256, 0, stream>>>(A_mid, Bm);
    pack_last_kernel<<<(NKC * FRAG + 255) / 256, 256, 0, stream>>>(A_last, Bl);
    chain_kernel<<<NSTILE, 256, 0, stream>>>(yA, Bf, Bm, Bl, out);
}
